// RoPEMaskedAttentionHead_34411277975722
// MI455X (gfx1250) — matrix operand order repaired and checked by reading, not compiled, not run on hardware
//
#include <hip/hip_runtime.h>
#include <hip/hip_bf16.h>
#include <math.h>

// ---------------------------------------------------------------------------
// RoPE masked attention head for MI455X (gfx1250, wave32, WMMA).
// bf16 operands into v_wmma_f32_16x16x32_bf16, f32 accumulation/softmax.
// V is produced TRANSPOSED by the projection GEMM so that every attention-side
// B fragment is a contiguous-per-lane load (merges into global_load_b128).
// Workspace (bf16): xb[16M] wq/wk/wv[1M each] qb/kb[16M each] vT[16M] ~140MB.
// ---------------------------------------------------------------------------

typedef __bf16 bf16_t;
typedef __attribute__((ext_vector_type(16))) __bf16 v16bf;
typedef __attribute__((ext_vector_type(8)))  float  v8f;
typedef __attribute__((ext_vector_type(4)))  int    v4i;

#define D_MODEL 1024
#define SEQ     4096
#define BATCH   4
#define NROWS   (BATCH * SEQ)       // 16384

#define HAVE_ASYNC_LDS                                             \
  (__has_builtin(__builtin_amdgcn_global_load_async_to_lds_b128) && \
   __has_builtin(__builtin_amdgcn_s_wait_asynccnt))

#if HAVE_ASYNC_LDS
typedef __attribute__((address_space(1))) v4i* gas_p;   // global int4*
typedef __attribute__((address_space(3))) v4i* lds_p;   // LDS int4*
#endif

// ---------------- WMMA fragment loaders (ISA 7.12.2 layouts) ----------------

// A matrix 16x32 bf16 (MxK), row-major source with leading dim ld.
// lanes 0-15: M=lane, regs0-3 K=0..7 (2/VGPR), regs4-7 K=16..23
// lanes16-31: M=lane-16, regs0-3 K=8..15,        regs4-7 K=24..31
__device__ inline v16bf load_A_frag(const bf16_t* base, int ld, int lane) {
  const int m  = lane & 15;
  const int kh = (lane >> 4) ? 8 : 0;
  const bf16_t* row = base + (size_t)m * ld;
  v16bf a;
#pragma unroll
  for (int v = 0; v < 4; ++v) {
    a[2 * v]     = row[kh + 2 * v];
    a[2 * v + 1] = row[kh + 2 * v + 1];
    a[8 + 2 * v]     = row[16 + kh + 2 * v];
    a[8 + 2 * v + 1] = row[16 + kh + 2 * v + 1];
  }
  return a;
}

// B matrix 32x16 bf16 (KxN) where memory holds rows of N with K contiguous:
// mem[n*ld + k] (w[n][k] for y = x W^T; K[j][d] for Q K^T; V^T[d][j] for P V).
// lanes 0-15: N=lane, K=0..7,16..23 (2/VGPR); lanes16-31: N=lane-16, K=8..15,24..31.
__device__ inline v16bf load_B_fragN(const bf16_t* base, int ld, int lane) {
  const int n  = lane & 15;
  const int kh = (lane >> 4) ? 8 : 0;
  const bf16_t* row = base + (size_t)n * ld + kh;
  v16bf b;
#pragma unroll
  for (int v = 0; v < 4; ++v) {
    b[2 * v]     = row[2 * v];
    b[2 * v + 1] = row[2 * v + 1];
    b[8 + 2 * v]     = row[16 + 2 * v];
    b[8 + 2 * v + 1] = row[16 + 2 * v + 1];
  }
  return b;
}

// ------------------------------- kernels ------------------------------------

__global__ void f32_to_bf16_kernel(const float* __restrict__ in,
                                   bf16_t* __restrict__ out, int n) {
  int i = blockIdx.x * blockDim.x + threadIdx.x;
  if (i < n) out[i] = (bf16_t)in[i];
}

// Out[m][n] = sum_k X[m][k] * W[n][k]   (nn.Linear, y = x W^T)
// grid = (Mtiles=1024, 2), block = 256 (8 waves). Each wave: 16x64 strip.
// transposeOut != 0: store Out^T per batch as [D_MODEL][SEQ] (used for V).
__global__ __launch_bounds__(256) void qkv_gemm_kernel(
    const bf16_t* __restrict__ X, const bf16_t* __restrict__ W,
    bf16_t* __restrict__ Out, int transposeOut) {
  __shared__ bf16_t sA[16 * 32];
  const int mtile = blockIdx.x;
  const int nhalf = blockIdx.y;
  const int tid  = threadIdx.x;
  const int wave = tid >> 5;
  const int lane = tid & 31;
  const int n0   = nhalf * 512 + wave * 64;

  v8f zf = {};
  v8f acc[4];
#pragma unroll
  for (int t = 0; t < 4; ++t) acc[t] = zf;

  const bf16_t* xrow = X + (size_t)mtile * 16 * D_MODEL;
  for (int k0 = 0; k0 < D_MODEL; k0 += 32) {
    // stage A tile 16x32 bf16 (1KB) into LDS
#if HAVE_ASYNC_LDS
    if (tid < 64) {  // 64 x 16B async chunks (GLOBAL_LOAD_ASYNC_TO_LDS_B128)
      const int row = tid >> 2, col = (tid & 3) * 8;
      const bf16_t* g = xrow + (size_t)row * D_MODEL + k0 + col;
      bf16_t* l = &sA[row * 32 + col];
      __builtin_amdgcn_global_load_async_to_lds_b128(
          (gas_p)(uintptr_t)g, (lds_p)(uint32_t)(uintptr_t)l, 0, 0);
    }
    __builtin_amdgcn_s_wait_asynccnt(0);
#else
    {
      int i = tid;
      sA[i] = xrow[(size_t)(i >> 5) * D_MODEL + (i & 31) + k0];
      int j = tid + 256;
      sA[j] = xrow[(size_t)(j >> 5) * D_MODEL + (j & 31) + k0];
    }
#endif
    __syncthreads();
    v16bf a = load_A_frag(sA, 32, lane);
#pragma unroll
    for (int t = 0; t < 4; ++t) {
      v16bf b = load_B_fragN(W + (size_t)(n0 + t * 16) * D_MODEL + k0,
                             D_MODEL, lane);
      acc[t] = __builtin_amdgcn_wmma_f32_16x16x32_bf16(
          false, a, false, b, (short)0, acc[t], false, false);
    }
    __syncthreads();
  }

  // C layout: reg e -> M = e + 8*(lane>=16), N = n0 + t*16 + (lane&15)
  const int row0 = (lane >> 4) << 3;
  const int col  = lane & 15;
  if (!transposeOut) {
    const int mrow = mtile * 16 + row0;
#pragma unroll
    for (int t = 0; t < 4; ++t) {
      bf16_t* o = Out + (size_t)mrow * D_MODEL + n0 + t * 16 + col;
#pragma unroll
      for (int e = 0; e < 8; ++e) o[(size_t)e * D_MODEL] = (bf16_t)acc[t][e];
    }
  } else {
    // Out^T[b][n][s]: per lane the 8 'e' elements are consecutive in s.
    const int b      = (mtile * 16) / SEQ;
    const int s_base = (mtile * 16) % SEQ + row0;
    bf16_t* Ob = Out + (size_t)b * D_MODEL * SEQ;
#pragma unroll
    for (int t = 0; t < 4; ++t) {
      bf16_t* o = Ob + (size_t)(n0 + t * 16 + col) * SEQ + s_base;
#pragma unroll
      for (int e = 0; e < 8; ++e) o[e] = (bf16_t)acc[t][e];
    }
  }
}

// In-place RoPE on Q and K (bf16), faithful theta = 10000^(-2(i-1)/d).
__global__ void rope_kernel(bf16_t* __restrict__ Q, bf16_t* __restrict__ K) {
  int idx = blockIdx.x * blockDim.x + threadIdx.x;  // over NROWS * d/2
  if (idx >= NROWS * (D_MODEL / 2)) return;
  int row = idx / (D_MODEL / 2);
  int i   = idx % (D_MODEL / 2);
  int m   = row % SEQ;
  float theta = __powf(10000.0f, -2.0f * ((float)i - 1.0f) / (float)D_MODEL);
  float ang = (float)m * theta;
  float s, c;
  sincosf(ang, &s, &c);
  size_t p = (size_t)row * D_MODEL + 2 * i;
  float qe = (float)Q[p], qo = (float)Q[p + 1];
  Q[p]     = (bf16_t)(qe * c + qo * s);
  Q[p + 1] = (bf16_t)(-qe * s + qo * c);
  float ke = (float)K[p], ko = (float)K[p + 1];
  K[p]     = (bf16_t)(ke * c + ko * s);
  K[p + 1] = (bf16_t)(-ke * s + ko * c);
}

// Flash attention, causal. grid = (SEQ/16, BATCH), block = 256 (8 waves).
// Each WG owns a 16-row Q block; wave w owns d-slice [w*128, w*128+128).
#define BM 16
#define BN 32
#define DW 128
__global__ __launch_bounds__(256) void attn_kernel(
    const bf16_t* __restrict__ Q, const bf16_t* __restrict__ K,
    const bf16_t* __restrict__ VT, float* __restrict__ O) {
  __shared__ float  sS[BM][BN];
  __shared__ bf16_t sP[BM][BN];
  __shared__ float  sM[BM], sL[BM], sAlpha[BM];

  const int qblk  = blockIdx.x;
  const int batch = blockIdx.y;
  const int tid  = threadIdx.x;
  const int wave = tid >> 5;
  const int lane = tid & 31;

  const bf16_t* Qb  = Q  + (size_t)batch * SEQ * D_MODEL;
  const bf16_t* Kb  = K  + (size_t)batch * SEQ * D_MODEL;
  const bf16_t* VTb = VT + (size_t)batch * D_MODEL * SEQ;  // [D_MODEL][SEQ]

  const int q0 = qblk * BM;
  const int d0 = wave * DW;

  // persistent Q fragments: this wave's 16 x 128 slice (4 k-steps of 32)
  v16bf qf[4];
#pragma unroll
  for (int ks = 0; ks < 4; ++ks)
    qf[ks] = load_A_frag(Qb + (size_t)q0 * D_MODEL + d0 + ks * 32,
                         D_MODEL, lane);

  v8f zf = {};
  v8f oacc[8];
#pragma unroll
  for (int t = 0; t < 8; ++t) oacc[t] = zf;

  if (tid < BM) { sM[tid] = -1e30f; sL[tid] = 0.0f; }

  const float scale = 0.03125f;  // 1/sqrt(1024)
  const int nkv = (q0 + BM - 1) / BN + 1;

  for (int jb = 0; jb < nkv; ++jb) {
    const int j0 = jb * BN;

    // zero the shared score block
    ((float*)sS)[tid]       = 0.0f;
    ((float*)sS)[tid + 256] = 0.0f;
    __syncthreads();

    if (jb + 1 < nkv) {  // prefetch next K/V block (global_prefetch_b8)
      __builtin_prefetch(Kb + (size_t)(j0 + BN + (lane & 15)) * D_MODEL + d0, 0, 0);
      __builtin_prefetch(VTb + (size_t)(d0 + (lane & 15)) * SEQ + j0 + BN, 0, 0);
    }

    // S partial = Q_slice @ K_slice^T over this wave's 128-wide d-slice
    v8f sacc[2];
    sacc[0] = zf; sacc[1] = zf;
#pragma unroll
    for (int ks = 0; ks < 4; ++ks) {
#pragma unroll
      for (int nt = 0; nt < 2; ++nt) {
        v16bf bk = load_B_fragN(
            Kb + (size_t)(j0 + nt * 16) * D_MODEL + d0 + ks * 32,
            D_MODEL, lane);
        sacc[nt] = __builtin_amdgcn_wmma_f32_16x16x32_bf16(
            false, qf[ks], false, bk, (short)0, sacc[nt], false, false);
      }
    }
    // reduce partials across the 8 waves into LDS (ds_add_f32)
    {
      const int row0 = (lane >> 4) << 3;
      const int col  = lane & 15;
#pragma unroll
      for (int nt = 0; nt < 2; ++nt)
#pragma unroll
        for (int e = 0; e < 8; ++e)
          atomicAdd(&sS[row0 + e][nt * 16 + col], sacc[nt][e]);
    }
    __syncthreads();

    // online softmax for the 16 rows (one thread per row)
    if (tid < BM) {
      const int gi = q0 + tid;
      float mprev = sM[tid];
      float mnew  = mprev;
      float sv[BN];
#pragma unroll
      for (int j = 0; j < BN; ++j) {
        float s = sS[tid][j] * scale;
        if (j0 + j > gi) s = -1e30f;  // causal mask
        sv[j] = s;
        mnew  = fmaxf(mnew, s);
      }
      float lsum = 0.0f;
#pragma unroll
      for (int j = 0; j < BN; ++j) {
        float p = __expf(sv[j] - mnew);
        sP[tid][j] = (bf16_t)p;
        lsum += p;
      }
      float alpha = __expf(mprev - mnew);
      sAlpha[tid] = alpha;
      sM[tid]     = mnew;
      sL[tid]     = sL[tid] * alpha + lsum;
    }
    __syncthreads();

    // rescale O and accumulate P @ V on this wave's d-slice.
    // B for PV comes from V^T[d][j]: contiguous per lane -> b128 loads.
    {
      const int row0 = (lane >> 4) << 3;
      float al[8];
#pragma unroll
      for (int e = 0; e < 8; ++e) al[e] = sAlpha[row0 + e];
      v16bf pf = load_A_frag(&sP[0][0], BN, lane);
#pragma unroll
      for (int t = 0; t < 8; ++t) {
#pragma unroll
        for (int e = 0; e < 8; ++e) oacc[t][e] *= al[e];
        v16bf bv = load_B_fragN(
            VTb + (size_t)(d0 + t * 16) * SEQ + j0, SEQ, lane);
        oacc[t] = __builtin_amdgcn_wmma_f32_16x16x32_bf16(
            false, pf, false, bv, (short)0, oacc[t], false, false);
      }
    }
    __syncthreads();
  }

  // epilogue: O / l  -> f32 output
  {
    const int row0 = (lane >> 4) << 3;
    const int col  = lane & 15;
    float rl[8];
#pragma unroll
    for (int e = 0; e < 8; ++e) rl[e] = 1.0f / sL[row0 + e];
    float* Ob = O + (size_t)batch * SEQ * D_MODEL;
#pragma unroll
    for (int t = 0; t < 8; ++t) {
      float* op = Ob + (size_t)(q0 + row0) * D_MODEL + d0 + t * 16 + col;
#pragma unroll
      for (int e = 0; e < 8; ++e)
        op[(size_t)e * D_MODEL] = oacc[t][e] * rl[e];
    }
  }
}

// ------------------------------ launcher ------------------------------------

extern "C" void kernel_launch(void* const* d_in, const int* in_sizes, int n_in,
                              void* d_out, int out_size, void* d_ws,
                              size_t ws_size, hipStream_t stream) {
  (void)in_sizes; (void)n_in; (void)out_size; (void)ws_size;

  const float* x  = (const float*)d_in[0];
  const float* wq = (const float*)d_in[1];
  const float* wk = (const float*)d_in[2];
  const float* wv = (const float*)d_in[3];
  float* out = (float*)d_out;

  const size_t NX = (size_t)BATCH * SEQ * D_MODEL;  // 16,777,216
  const size_t NW = (size_t)D_MODEL * D_MODEL;      // 1,048,576

  char* ws = (char*)d_ws;
  bf16_t* xb  = (bf16_t*)ws; ws += NX * sizeof(bf16_t);
  bf16_t* wqb = (bf16_t*)ws; ws += NW * sizeof(bf16_t);
  bf16_t* wkb = (bf16_t*)ws; ws += NW * sizeof(bf16_t);
  bf16_t* wvb = (bf16_t*)ws; ws += NW * sizeof(bf16_t);
  bf16_t* qb  = (bf16_t*)ws; ws += NX * sizeof(bf16_t);
  bf16_t* kb  = (bf16_t*)ws; ws += NX * sizeof(bf16_t);
  bf16_t* vbT = (bf16_t*)ws; ws += NX * sizeof(bf16_t);  // per-batch [D][SEQ]

  const int thr = 256;
  f32_to_bf16_kernel<<<(int)((NX + thr - 1) / thr), thr, 0, stream>>>(x, xb, (int)NX);
  f32_to_bf16_kernel<<<(int)((NW + thr - 1) / thr), thr, 0, stream>>>(wq, wqb, (int)NW);
  f32_to_bf16_kernel<<<(int)((NW + thr - 1) / thr), thr, 0, stream>>>(wk, wkb, (int)NW);
  f32_to_bf16_kernel<<<(int)((NW + thr - 1) / thr), thr, 0, stream>>>(wv, wvb, (int)NW);

  dim3 ggrid(NROWS / 16, 2);
  qkv_gemm_kernel<<<ggrid, 256, 0, stream>>>(xb, wqb, qb, 0);
  qkv_gemm_kernel<<<ggrid, 256, 0, stream>>>(xb, wkb, kb, 0);
  qkv_gemm_kernel<<<ggrid, 256, 0, stream>>>(xb, wvb, vbT, 1);

  const int nrope = NROWS * (D_MODEL / 2);
  rope_kernel<<<(nrope + thr - 1) / thr, thr, 0, stream>>>(qb, kb);

  dim3 agrid(SEQ / BM, BATCH);
  attn_kernel<<<agrid, 256, 0, stream>>>(qb, kb, vbT, out);
}
